// RelativeEncoder_35270271434918
// MI455X (gfx1250) — matrix operand order repaired and checked by reading, not compiled, not run on hardware
//
#include <hip/hip_runtime.h>
#include <math.h>

#define N_NODES   100000
#define N_EDGES   1600000
#define NODE_F    25
#define HID       128
#define LN_EPS    1e-5f

typedef __attribute__((ext_vector_type(16))) _Float16 v16h;
typedef __attribute__((ext_vector_type(8)))  _Float16 v8h;
typedef __attribute__((ext_vector_type(8)))  float    v8f;

// ---------------- LDS layout (dynamic shared) ----------------
// [0, 8K)            w1T  : [128][32]  f16  (layer1 weights, transposed, K padded)
// [8K, 40K)          w2T  : [128][128] f16
// [40K, 72K)         w3T  : [128][128] f16
// [72K, 72K+2.5K)    params: b1,b2,b3,g,beta (f32, 128 each)
// [76288, +8*PW)     per-wave scratch, PW bytes each:
//    +0     in0  : [16][32]  f16   (layer-1 act source, row-major)
//    +1024  bufA : [16][128] f16   (layer-1 out / layer-2 act source)
//    +5120  bufB : [16][128] f16   (layer-2 out / layer-3 act source)
//    +0     hF   : [16][128] f32   (layer-3 post-bias rows; overlaps the f16
//                                   buffers — all dead once layer-3 B-frags
//                                   are register-resident)
#define W1T_OFF   0
#define W2T_OFF   8192
#define W3T_OFF   40960
#define PRM_OFF   73728
#define PB1_OFF   (PRM_OFF + 0)
#define PB2_OFF   (PRM_OFF + 512)
#define PB3_OFF   (PRM_OFF + 1024)
#define PG_OFF    (PRM_OFF + 1536)
#define PBT_OFF   (PRM_OFF + 2048)
#define PWV_OFF   76288
#define PW        9216
#define SMEM_BYTES (PWV_OFF + 8 * PW)

// Single-instruction ReLU: v_med3_num_f32(x, 0, +inf) — avoids the extra
// canonicalization max that fmaxf(x, 0) lowers to.
__device__ __forceinline__ float relu1(float x) {
    return __builtin_amdgcn_fmed3f(x, 0.0f, __builtin_inff());
}

// D = A x B with A = W^T (16 out-neurons x 32 K), B = act^T (32 K x 16 rows),
// C preloaded with the bias (WMMA computes A*B + C, so bias-add is free).
//
// A-fragment (16-bit A 16x32, contiguous k order as in B): lane l holds A-row i = l&15 (neuron t*16+i);
// elem e -> K = 16*(l>>4)+e.
__device__ __forceinline__ v16h load_w_frag(const _Float16* wT, int ldk, int ntile,
                                            int kbase, int lane) {
    const _Float16* p = wT + (ntile * 16 + (lane & 15)) * ldk + kbase + (lane >> 4) * 16;
    return *(const v16h*)(p);
}

// B-fragment (16-bit B 32x16): lane l holds B-col = act row m = l&15;
// elem e -> K = 16*(l>>4)+e. Row-major act => one contiguous aligned 32B read.
__device__ __forceinline__ v16h load_act_frag(const _Float16* act, int ld,
                                              int kbase, int lane) {
    return *(const v16h*)(act + (lane & 15) * ld + kbase + (lane >> 4) * 16);
}

// Hidden layer for one 16-row tile: actOut = relu(actIn @ W + b).
// D-layout: lane l holds row m = l&15, neurons n = t*16 + 8*(l>>4) + r (contiguous).
template <int NK>
__device__ __forceinline__ void mlp_layer(const _Float16* actIn, int ldIn,
                                          const _Float16* wT,
                                          const float* biasLds,   // LDS-resident
                                          _Float16* actOut, int lane) {
    v16h bfrag[NK];
#pragma unroll
    for (int c = 0; c < NK; ++c) bfrag[c] = load_act_frag(actIn, ldIn, c * 32, lane);
    __builtin_amdgcn_wave_barrier();

    const int m = lane & 15, h = lane >> 4;
    const int ldk = NK * 32;
#pragma unroll
    for (int t = 0; t < 8; ++t) {
        v8f acc = *(const v8f*)(biasLds + t * 16 + h * 8);   // bias as WMMA C input
#pragma unroll
        for (int c = 0; c < NK; ++c) {
            v16h a = load_w_frag(wT, ldk, t, c * 32, lane);
            acc = __builtin_amdgcn_wmma_f32_16x16x32_f16(
                false, a, false, bfrag[c], (short)0, acc, false, false);
        }
        v8h o;
#pragma unroll
        for (int r = 0; r < 8; ++r)
            o[r] = (_Float16)relu1(acc[r]);
        *(v8h*)(actOut + m * HID + t * 16 + h * 8) = o;      // one ds_store_b128
    }
    __builtin_amdgcn_wave_barrier();
}

// Final layer + fused LayerNorm. Post-bias rows spill to the LDS f32 buffer
// (keeps register pressure low); stats stay in regs + one shfl_xor.
__device__ __forceinline__ void mlp_final_ln(const _Float16* actIn,
                                             const _Float16* w3T,
                                             const float* b3Lds,
                                             const float* gLds,
                                             const float* btLds,
                                             float* hF,              // LDS [16][128] f32
                                             float* __restrict__ outRow0,
                                             int validRows, int lane) {
    v16h bfrag[4];
#pragma unroll
    for (int c = 0; c < 4; ++c) bfrag[c] = load_act_frag(actIn, HID, c * 32, lane);
    __builtin_amdgcn_wave_barrier();   // all f16 buffers dead -> hF may overwrite

    const int m = lane & 15, h = lane >> 4;
    float s = 0.0f, sq = 0.0f;
#pragma unroll
    for (int t = 0; t < 8; ++t) {
        v8f acc = *(const v8f*)(b3Lds + t * 16 + h * 8);     // bias as WMMA C input
#pragma unroll
        for (int c = 0; c < 4; ++c) {
            v16h a = load_w_frag(w3T, HID, t, c * 32, lane);
            acc = __builtin_amdgcn_wmma_f32_16x16x32_f16(
                false, a, false, bfrag[c], (short)0, acc, false, false);
        }
#pragma unroll
        for (int r = 0; r < 8; ++r) {
            s += acc[r]; sq += acc[r] * acc[r];
        }
        *(v8f*)(hF + m * HID + t * 16 + h * 8) = acc;        // 2x ds_store_b128
    }
    s  += __shfl_xor(s, 16, 32);     // lane pair (l, l^16) covers the full row
    sq += __shfl_xor(sq, 16, 32);
    const float mean = s * (1.0f / HID);
    const float inv  = rsqrtf(sq * (1.0f / HID) - mean * mean + LN_EPS);

    if (m < validRows) {
        float* orow = outRow0 + (long long)m * HID;
#pragma unroll
        for (int t = 0; t < 8; ++t) {
            const v8f hv = *(const v8f*)(hF + m * HID + t * 16 + h * 8);
            const v8f gv = *(const v8f*)(gLds  + t * 16 + h * 8);
            const v8f bb = *(const v8f*)(btLds + t * 16 + h * 8);
            v8f o;
#pragma unroll
            for (int r = 0; r < 8; ++r)
                o[r] = (hv[r] - mean) * inv * gv[r] + bb[r];
            *(v8f*)(orow + t * 16 + h * 8) = o;     // 32B store; lane pairs -> 64B runs
        }
    }
}

__device__ __forceinline__ void stage_weights_params(char* smem,
                                                     const float* __restrict__ w1, int fin1,
                                                     const float* __restrict__ w2,
                                                     const float* __restrict__ w3,
                                                     const float* __restrict__ b1,
                                                     const float* __restrict__ b2,
                                                     const float* __restrict__ b3,
                                                     const float* __restrict__ g,
                                                     const float* __restrict__ beta,
                                                     int tid, int nth) {
    _Float16* w1T = (_Float16*)(smem + W1T_OFF);
    _Float16* w2T = (_Float16*)(smem + W2T_OFF);
    _Float16* w3T = (_Float16*)(smem + W3T_OFF);
    for (int i = tid; i < HID * 32; i += nth) {
        int n = i >> 5, k = i & 31;
        w1T[i] = (k < fin1) ? (_Float16)w1[k * HID + n] : (_Float16)0.0f;
    }
    for (int i = tid; i < HID * HID; i += nth) {
        int n = i >> 7, k = i & 127;
        w2T[i] = (_Float16)w2[k * HID + n];
        w3T[i] = (_Float16)w3[k * HID + n];
    }
    float* pb1 = (float*)(smem + PB1_OFF);
    float* pb2 = (float*)(smem + PB2_OFF);
    float* pb3 = (float*)(smem + PB3_OFF);
    float* pg  = (float*)(smem + PG_OFF);
    float* pbt = (float*)(smem + PBT_OFF);
    for (int i = tid; i < HID; i += nth) {
        pb1[i] = b1[i]; pb2[i] = b2[i]; pb3[i] = b3[i];
        pg[i]  = g[i];  pbt[i] = beta[i];
    }
}

__device__ __forceinline__ void run_tile(char* smem, char* mybase, int lane,
                                         float* outRow0, int validRows) {
    _Float16* in0  = (_Float16*)(mybase);
    _Float16* bufA = (_Float16*)(mybase + 1024);
    _Float16* bufB = (_Float16*)(mybase + 5120);
    float*    hF   = (float*)(mybase);
    mlp_layer<1>(in0,  32,  (const _Float16*)(smem + W1T_OFF),
                 (const float*)(smem + PB1_OFF), bufA, lane);
    mlp_layer<4>(bufA, HID, (const _Float16*)(smem + W2T_OFF),
                 (const float*)(smem + PB2_OFF), bufB, lane);
    mlp_final_ln(bufB, (const _Float16*)(smem + W3T_OFF),
                 (const float*)(smem + PB3_OFF),
                 (const float*)(smem + PG_OFF),
                 (const float*)(smem + PBT_OFF),
                 hF, outRow0, validRows, lane);
}

// ---------------- Node kernel (grid-stride over 16-row tiles) ----------------
__global__ void node_mlp_kernel(const float* __restrict__ x,
                                const float* __restrict__ w1, const float* __restrict__ b1,
                                const float* __restrict__ w2, const float* __restrict__ b2,
                                const float* __restrict__ w3, const float* __restrict__ b3,
                                const float* __restrict__ g,  const float* __restrict__ beta,
                                float* __restrict__ out) {
    extern __shared__ __align__(16) char smem[];
    const int tid = threadIdx.x;
    stage_weights_params(smem, w1, NODE_F, w2, w3, b1, b2, b3, g, beta, tid, blockDim.x);
    __syncthreads();

    const int wave = tid >> 5, lane = tid & 31;
    char* mybase = smem + PWV_OFF + wave * PW;
    _Float16* in0 = (_Float16*)mybase;

    const int nTiles = (N_NODES + 15) / 16;
    for (int tile = blockIdx.x * 8 + wave; tile < nTiles; tile += gridDim.x * 8) {
        const int row0 = tile * 16;
        if (lane < 16) {                         // lanes 0..15 stage one row each
            const int r = row0 + lane;
            _Float16* dst = in0 + lane * 32;
            if (r < N_NODES) {
                const float* src = x + (long long)r * NODE_F;
#pragma unroll
                for (int k = 0; k < NODE_F; ++k) dst[k] = (_Float16)src[k];
#pragma unroll
                for (int k = NODE_F; k < 32; ++k) dst[k] = (_Float16)0.0f;
            } else {
#pragma unroll
                for (int k = 0; k < 32; ++k) dst[k] = (_Float16)0.0f;
            }
        }
        __builtin_amdgcn_wave_barrier();

        int valid = N_NODES - row0;
        if (valid > 16) valid = 16;
        run_tile(smem, mybase, lane, out + (long long)row0 * HID, valid);
    }
}

// ---------------- Edge kernel (grid-stride over 16-edge tiles) ----------------
__global__ void edge_mlp_kernel(const float* __restrict__ x,
                                const long long* __restrict__ ei, // [2][N_EDGES]
                                const float* __restrict__ w1, const float* __restrict__ b1,
                                const float* __restrict__ w2, const float* __restrict__ b2,
                                const float* __restrict__ w3, const float* __restrict__ b3,
                                const float* __restrict__ g,  const float* __restrict__ beta,
                                float* __restrict__ out) {
    extern __shared__ __align__(16) char smem[];
    const int tid = threadIdx.x;
    stage_weights_params(smem, w1, 4, w2, w3, b1, b2, b3, g, beta, tid, blockDim.x);
    __syncthreads();

    const int wave = tid >> 5, lane = tid & 31;
    char* mybase = smem + PWV_OFF + wave * PW;
    _Float16* in0 = (_Float16*)mybase;

    const int nTiles = (N_EDGES + 15) / 16;
    for (int tile = blockIdx.x * 8 + wave; tile < nTiles; tile += gridDim.x * 8) {
        const long long e0 = (long long)tile * 16;
        if (lane < 16) {                         // lanes 0..15 build one feature row each
            const long long e = e0 + lane;
            _Float16* dst = in0 + lane * 32;
            if (e < N_EDGES) {
                const long long r = ei[e];
                const long long c = ei[(long long)N_EDGES + e];
                const float* xr = x + r * NODE_F;
                const float* xc = x + c * NODE_F;
                const float d0 = xr[0] - xc[0];
                const float d1 = xr[1] - xc[1];
                const float d2 = xr[2] - xc[2];
                const float mag = sqrtf(d0 * d0 + d1 * d1 + d2 * d2);
                dst[0] = (_Float16)d0; dst[1] = (_Float16)d1;
                dst[2] = (_Float16)d2; dst[3] = (_Float16)mag;
#pragma unroll
                for (int k = 4; k < 32; ++k) dst[k] = (_Float16)0.0f;
            } else {
#pragma unroll
                for (int k = 0; k < 32; ++k) dst[k] = (_Float16)0.0f;
            }
        }
        __builtin_amdgcn_wave_barrier();

        long long validL = (long long)N_EDGES - e0;
        int valid = validL > 16 ? 16 : (int)validL;
        run_tile(smem, mybase, lane, out + e0 * HID, valid);
    }
}

extern "C" void kernel_launch(void* const* d_in, const int* in_sizes, int n_in,
                              void* d_out, int out_size, void* d_ws, size_t ws_size,
                              hipStream_t stream) {
    (void)in_sizes; (void)n_in; (void)d_ws; (void)ws_size; (void)out_size;

    const float*     x  = (const float*)d_in[0];
    const long long* ei = (const long long*)d_in[1];  // int64 [2, N_EDGES]
    const float* ew1 = (const float*)d_in[2];  const float* eb1 = (const float*)d_in[3];
    const float* ew2 = (const float*)d_in[4];  const float* eb2 = (const float*)d_in[5];
    const float* ew3 = (const float*)d_in[6];  const float* eb3 = (const float*)d_in[7];
    const float* eg  = (const float*)d_in[8];  const float* ebt = (const float*)d_in[9];
    const float* nw1 = (const float*)d_in[10]; const float* nb1 = (const float*)d_in[11];
    const float* nw2 = (const float*)d_in[12]; const float* nb2 = (const float*)d_in[13];
    const float* nw3 = (const float*)d_in[14]; const float* nb3 = (const float*)d_in[15];
    const float* ng  = (const float*)d_in[16]; const float* nbt = (const float*)d_in[17];

    float* node_out = (float*)d_out;                       // [N_NODES, 128]
    float* edge_out = node_out + (size_t)N_NODES * HID;    // [N_EDGES, 128]

    const int nodeBlocks = 782;    // 6250 tiles / 8 waves -> ~1 tile per wave
    const int edgeBlocks = 2500;   // 100000 tiles -> 5 tiles per wave

    node_mlp_kernel<<<nodeBlocks, 256, SMEM_BYTES, stream>>>(
        x, nw1, nb1, nw2, nb2, nw3, nb3, ng, nbt, node_out);
    edge_mlp_kernel<<<edgeBlocks, 256, SMEM_BYTES, stream>>>(
        x, ei, ew1, eb1, ew2, eb2, ew3, eb3, eg, ebt, edge_out);
}
